// DiagonalSSMLayer_62302795596624
// MI455X (gfx1250) — matrix operand order repaired and checked by reading, not compiled, not run on hardware
//
#include <hip/hip_runtime.h>
#include <hip/hip_bf16.h>

#define BB 8
#define LL 2048
#define DD 1024
#define SS 256
#define BL (BB * LL)   // 16384 rows

typedef __attribute__((ext_vector_type(16))) __bf16        v16bf;
typedef __attribute__((ext_vector_type(8)))  float         v8f;
typedef __attribute__((ext_vector_type(4)))  unsigned int  v4u;

// Load a 16-element bf16 WMMA fragment as two aligned 16-byte chunks at
// element offsets 0 and `second_off` from p.
__device__ __forceinline__ v16bf load_frag(const __bf16* p, int second_off) {
    union { v16bf v; v4u q[2]; } u;
    u.q[0] = *(const v4u*)(p);
    u.q[1] = *(const v4u*)(p + second_off);
    return u.v;
}

__device__ __forceinline__ v8f wmma_bf16(v16bf a, v16bf b, v8f c) {
    return __builtin_amdgcn_wmma_f32_16x16x32_bf16(
        /*neg_a=*/false, a, /*neg_b=*/false, b,
        /*c_mod=*/(short)0, c, /*reuse_a=*/false, /*reuse_b=*/false);
}

// ---------------------------------------------------------------------------
// Kernel 0: f32 -> bf16 weight conversion (runs once per launch; tiny).
// ---------------------------------------------------------------------------
__global__ void cvt_bf16_kernel(const float* __restrict__ src,
                                __bf16* __restrict__ dst, int n) {
    int i = blockIdx.x * blockDim.x + threadIdx.x;
    if (i < n) dst[i] = (__bf16)src[i];
}

// ---------------------------------------------------------------------------
// Kernel 1: fused projection GEMMs.
//   alpha = sigmoid(x @ Wg^T + bg), u = x @ WB^T + bB   (row = b*L + l)
// Workgroup = 256 threads (8 waves). Each block owns 16 rows x all 256 cols,
// so x is read from HBM exactly once. x tile (16x32) staged per k-step into
// a double-buffered LDS tile (one barrier per step), row stride padded to 40
// halfwords -> the two per-lane 16B fragment reads are bank-conflict free.
// ---------------------------------------------------------------------------
__global__ __launch_bounds__(256) void proj_kernel(
    const float*  __restrict__ x,
    const __bf16* __restrict__ Wgb,
    const __bf16* __restrict__ WBb,
    const float*  __restrict__ bg,
    const float*  __restrict__ bB,
    float* __restrict__ alphaF,
    float* __restrict__ uF) {

    __shared__ __bf16 sA[2][16 * 40];

    const int t    = threadIdx.x;
    const int lane = t & 31;
    const int wave = t >> 5;
    const int ln   = lane & 15;          // A: row M / B: col N / C: col N
    const int half = lane >> 4;          // 0 or 1
    const int kbA  = half * 16;          // A-fragment K base within tile
    const int kbB  = half * 16;          // B-fragment K base within tile

    const int row0 = blockIdx.x * 16;
    const int s0   = (wave * 2 + 0) * 16;
    const int s1   = (wave * 2 + 1) * 16;

    // staging assignment: thread t copies 2 consecutive elements
    const int sr = t >> 4;               // 0..15  (row of tile)
    const int sk = (t & 15) * 2;         // 0..30  (even k within tile)
    const float* xrow = x + (row0 + sr) * DD + sk;

    v8f accg0 = {}, accg1 = {}, accu0 = {}, accu1 = {};

    // register-held prefetch of the first x tile
    float2 xf = *(const float2*)(xrow);

    for (int k0 = 0; k0 < DD; k0 += 32) {
        const int p = (k0 >> 5) & 1;
        float2 cur = xf;
        if (k0 + 32 < DD)                 // prefetch next tile during compute
            xf = *(const float2*)(xrow + k0 + 32);

        __bf16* sbuf = &sA[p][0];
        sbuf[sr * 40 + sk + 0] = (__bf16)cur.x;
        sbuf[sr * 40 + sk + 1] = (__bf16)cur.y;
        __syncthreads();                  // single barrier per k-step

        // ---- A fragment from LDS (two ds_load_b128, conflict-free) ----
        v16bf afrag = load_frag(&sbuf[ln * 40 + kbA], 8);

        // ---- B fragments straight from bf16 weights (L2-resident) ----
        v16bf bgf0 = load_frag(Wgb + (s0 + ln) * DD + k0 + kbB, 8);
        v16bf bgf1 = load_frag(Wgb + (s1 + ln) * DD + k0 + kbB, 8);
        v16bf buf0 = load_frag(WBb + (s0 + ln) * DD + k0 + kbB, 8);
        v16bf buf1 = load_frag(WBb + (s1 + ln) * DD + k0 + kbB, 8);

        accg0 = wmma_bf16(afrag, bgf0, accg0);
        accg1 = wmma_bf16(afrag, bgf1, accg1);
        accu0 = wmma_bf16(afrag, buf0, accu0);
        accu1 = wmma_bf16(afrag, buf1, accu1);
    }

    // ---- epilogue: bias + sigmoid, store f32 ----
    const float bg0v = bg[s0 + ln], bg1v = bg[s1 + ln];
    const float bb0v = bB[s0 + ln], bb1v = bB[s1 + ln];

#pragma unroll
    for (int r = 0; r < 8; ++r) {
        const int row = row0 + r + half * 8;
        float g0 = accg0[r] + bg0v;
        float g1 = accg1[r] + bg1v;
        alphaF[row * SS + s0 + ln] = 1.0f / (1.0f + expf(-g0));
        alphaF[row * SS + s1 + ln] = 1.0f / (1.0f + expf(-g1));
        uF[row * SS + s0 + ln] = accu0[r] + bb0v;
        uF[row * SS + s1 + ln] = accu1[r] + bb1v;
    }
}

// ---------------------------------------------------------------------------
// Kernel 2: diagonal selective scan. One thread per (b, s) state.
// Loads across l are independent of the FMA chain: batch 16 steps of loads
// (32 outstanding loads/thread) ahead of the serial h = a*h + u chain so the
// 64 resident waves stay bandwidth- rather than latency-bound.
// ---------------------------------------------------------------------------
__global__ __launch_bounds__(256) void scan_kernel(
    const float* __restrict__ alphaF,
    const float* __restrict__ uF,
    __bf16* __restrict__ hsb) {

    const int t = blockIdx.x * blockDim.x + threadIdx.x;  // 0..2047
    const int b = t >> 8;
    const int s = t & 255;
    const int base = b * LL * SS + s;

    float h = 0.0f;
    for (int l0 = 0; l0 < LL; l0 += 16) {
        float a[16], u[16];
#pragma unroll
        for (int j = 0; j < 16; ++j) {
            const int idx = base + (l0 + j) * SS;
            a[j] = alphaF[idx];
            u[j] = uF[idx];
        }
#pragma unroll
        for (int j = 0; j < 16; ++j) {
            h = fmaf(a[j], h, u[j]);                 // h = a*h + u
            hsb[base + (l0 + j) * SS] = (__bf16)h;
        }
    }
}

// ---------------------------------------------------------------------------
// Kernel 3: output GEMM  y[row, d] = sum_s hs[row, s] * WC[d, s] + bC[d].
// Block = 8 waves: 16 rows x 128 cols; grid.y spans D/128 column groups.
// Same double-buffered LDS staging as proj_kernel (hs already bf16).
// ---------------------------------------------------------------------------
__global__ __launch_bounds__(256) void out_kernel(
    const __bf16* __restrict__ hsb,
    const __bf16* __restrict__ WCb,
    const float*  __restrict__ bC,
    float* __restrict__ y) {

    __shared__ __bf16 sA[2][16 * 40];

    const int t    = threadIdx.x;
    const int lane = t & 31;
    const int wave = t >> 5;
    const int ln   = lane & 15;
    const int half = lane >> 4;
    const int kbA  = half * 16;
    const int kbB  = half * 16;

    const int row0 = blockIdx.x * 16;
    const int d0   = blockIdx.y * 128 + wave * 16;

    const int sr = t >> 4;
    const int sk = (t & 15) * 2;
    const __bf16* hrow = hsb + (row0 + sr) * SS + sk;

    v8f acc = {};

    unsigned int stg = *(const unsigned int*)(hrow);   // prefetch first tile

    for (int k0 = 0; k0 < SS; k0 += 32) {
        const int p = (k0 >> 5) & 1;
        unsigned int cur = stg;
        if (k0 + 32 < SS)
            stg = *(const unsigned int*)(hrow + k0 + 32);

        __bf16* sbuf = &sA[p][0];
        *(unsigned int*)(&sbuf[sr * 40 + sk]) = cur;
        __syncthreads();

        v16bf afrag = load_frag(&sbuf[ln * 40 + kbA], 8);
        v16bf bfrag = load_frag(WCb + (d0 + ln) * SS + k0 + kbB, 8);
        acc = wmma_bf16(afrag, bfrag, acc);
    }

    const float bcv = bC[d0 + ln];
#pragma unroll
    for (int r = 0; r < 8; ++r) {
        const int row = row0 + r + half * 8;
        y[row * DD + d0 + ln] = acc[r] + bcv;
    }
}

// ---------------------------------------------------------------------------
extern "C" void kernel_launch(void* const* d_in, const int* in_sizes, int n_in,
                              void* d_out, int out_size, void* d_ws, size_t ws_size,
                              hipStream_t stream) {
    const float* x  = (const float*)d_in[0];
    const float* Wg = (const float*)d_in[1];
    const float* bg = (const float*)d_in[2];
    const float* WB = (const float*)d_in[3];
    const float* bB = (const float*)d_in[4];
    const float* WC = (const float*)d_in[5];
    const float* bC = (const float*)d_in[6];
    float* yOut = (float*)d_out;

    // ---- workspace layout ----
    char* ws = (char*)d_ws;
    size_t off = 0;
    float*  alphaF = (float*)(ws + off);  off += (size_t)BL * SS * sizeof(float);   // 16 MB
    float*  uF     = (float*)(ws + off);  off += (size_t)BL * SS * sizeof(float);   // 16 MB
    __bf16* hsb    = (__bf16*)(ws + off); off += (size_t)BL * SS * sizeof(__bf16);  //  8 MB
    __bf16* Wgb    = (__bf16*)(ws + off); off += (size_t)SS * DD * sizeof(__bf16);
    __bf16* WBb    = (__bf16*)(ws + off); off += (size_t)SS * DD * sizeof(__bf16);
    __bf16* WCb    = (__bf16*)(ws + off); off += (size_t)DD * SS * sizeof(__bf16);

    const int nW = SS * DD;  // 262144 for each weight matrix
    cvt_bf16_kernel<<<(nW + 255) / 256, 256, 0, stream>>>(Wg, Wgb, nW);
    cvt_bf16_kernel<<<(nW + 255) / 256, 256, 0, stream>>>(WB, WBb, nW);
    cvt_bf16_kernel<<<(nW + 255) / 256, 256, 0, stream>>>(WC, WCb, nW);

    // projection: 1024 blocks x 256 threads (16 rows x 256 cols each)
    proj_kernel<<<BL / 16, 256, 0, stream>>>(x, Wgb, WBb, bg, bB, alphaF, uF);

    // scan: one thread per (b, s)
    scan_kernel<<<(BB * SS) / 256, 256, 0, stream>>>(alphaF, uF, hsb);

    // output GEMM: 1024 row-tiles x 8 col-groups
    out_kernel<<<dim3(BL / 16, DD / 128), 256, 0, stream>>>(hsb, WCb, bC, yOut);
}
